// a3d_module_22617297781392
// MI455X (gfx1250) — compile-verified, matrix operand order checked by reading, not run on hardware
//
#include <hip/hip_runtime.h>
#include <hip/hip_bf16.h>

// ---------------------------------------------------------------------------
// Problem constants (from reference): B=4, CIN=512, C=128, N=8*28*28=6272
// ---------------------------------------------------------------------------
#define NPOS   6272
#define NBATCH 4
#define CINCH  512
#define CRED   128

typedef __attribute__((ext_vector_type(16))) __bf16 v16bf;
typedef __attribute__((ext_vector_type(8)))  __bf16 v8bf;
typedef __attribute__((ext_vector_type(8)))  float  v8f;

// Load a 16x32 bf16 A/B fragment's per-lane 16 elements.
// Per CDNA5 ISA layout: lane row = lane%16 ; elements 0..7 hold K = hi8+0..7,
// elements 8..15 hold K = 16+hi8+0..7 (hi8 = (lane>=16)?8:0).
// Caller passes p = rowBase + chunkBase + hi8  (16B aligned).
static __device__ __forceinline__ v16bf load_frag(const __bf16* __restrict__ p) {
  v8bf lo = *(const v8bf*)(p);
  v8bf hi = *(const v8bf*)(p + 16);
  v16bf f;
#pragma unroll
  for (int t = 0; t < 8; ++t) { f[t] = lo[t]; f[t + 8] = hi[t]; }
  return f;
}

static __device__ __forceinline__ v8f wmma_bf16(v16bf a, v16bf b, v8f c) {
  return __builtin_amdgcn_wmma_f32_16x16x32_bf16(false, a, false, b,
                                                 (short)0, c, false, false);
}

// ---------------------------------------------------------------------------
// Kernel 1: fp32 -> bf16 weight conversion (k_w, q_w, v_w are 128x512,
// r_w is 512x128 — all 65536 elements).
// ---------------------------------------------------------------------------
__global__ void convert_weights(const float* __restrict__ kw, const float* __restrict__ qw,
                                const float* __restrict__ vw, const float* __restrict__ rw,
                                __bf16* __restrict__ kwb, __bf16* __restrict__ qwb,
                                __bf16* __restrict__ vwb, __bf16* __restrict__ rwb) {
  int i = blockIdx.x * blockDim.x + threadIdx.x;
  if (i < CRED * CINCH) {
    kwb[i] = (__bf16)kw[i];
    qwb[i] = (__bf16)qw[i];
    vwb[i] = (__bf16)vw[i];
    rwb[i] = (__bf16)rw[i];
  }
}

// ---------------------------------------------------------------------------
// Kernel 2: x [B, CIN, N] fp32  ->  xT [B, N, CIN] bf16 (LDS-tiled transpose)
// grid(196, 16, 4), block(32, 8)
// ---------------------------------------------------------------------------
__global__ void transpose_convert_x(const float* __restrict__ x, __bf16* __restrict__ xT) {
  __shared__ float tile[32][33];
  const int b  = blockIdx.z;
  const int n0 = blockIdx.x * 32;
  const int c0 = blockIdx.y * 32;
  const int tx = threadIdx.x;   // 0..31
  const int ty = threadIdx.y;   // 0..7
  const float* xb = x + (size_t)b * CINCH * NPOS;
#pragma unroll
  for (int k = 0; k < 32; k += 8)
    tile[ty + k][tx] = xb[(size_t)(c0 + ty + k) * NPOS + (n0 + tx)];
  __syncthreads();
  __bf16* xTb = xT + (size_t)b * NPOS * CINCH;
#pragma unroll
  for (int k = 0; k < 32; k += 8)
    xTb[(size_t)(n0 + ty + k) * CINCH + (c0 + tx)] = (__bf16)tile[tx][ty + k];
}

// ---------------------------------------------------------------------------
// Kernel 3: projections.  One wave computes a 16-position x 128-channel tile
// for one of {K, Q, V}.   K, Q stored position-major [B, N, 128] bf16;
// V stored channel-major  [B, 128, N] bf16 (so attention V B-frags load
// contiguously along the j dimension).
// All 9 fragment loads of a K-step are issued as one clause before the 8
// WMMAs so the waits are partial instead of serial full stalls.
// grid(392, 3, 4), block(32)
// ---------------------------------------------------------------------------
__global__ void proj_kqv(const __bf16* __restrict__ xT,
                         const __bf16* __restrict__ kwb, const __bf16* __restrict__ qwb,
                         const __bf16* __restrict__ vwb,
                         const float* __restrict__ kb, const float* __restrict__ qb,
                         const float* __restrict__ vb,
                         __bf16* __restrict__ Kp, __bf16* __restrict__ Qp,
                         __bf16* __restrict__ Vt) {
  const int lane = threadIdx.x;
  const int row  = lane & 15;
  const int hi8  = (lane >> 4) * 8;
  const int i0   = blockIdx.x * 16;
  const int proj = blockIdx.y;
  const int b    = blockIdx.z;

  const __bf16* w    = (proj == 0) ? kwb : (proj == 1) ? qwb : vwb;
  const float*  bias = (proj == 0) ? kb  : (proj == 1) ? qb  : vb;

  const __bf16* xrow = xT + ((size_t)b * NPOS + (i0 + row)) * CINCH + hi8;
  const __bf16* wrow = w + (size_t)row * CINCH + hi8;

  v8f acc[8];
#pragma unroll
  for (int ob = 0; ob < 8; ++ob) acc[ob] = (v8f){};

  for (int kk = 0; kk < CINCH / 32; ++kk) {            // 16 K-steps of 32
    // Issue all loads first (one clause), then consume with WMMAs.
    v16bf a = load_frag(xrow + kk * 32);
    v16bf bf[8];
#pragma unroll
    for (int ob = 0; ob < 8; ++ob)
      bf[ob] = load_frag(wrow + (size_t)(ob * 16) * CINCH + kk * 32);
#pragma unroll
    for (int ob = 0; ob < 8; ++ob)
      acc[ob] = wmma_bf16(a, bf[ob], acc[ob]);
  }

  if (proj < 2) {
    __bf16* out = ((proj == 0) ? Kp : Qp) + (size_t)b * NPOS * CRED;
#pragma unroll
    for (int ob = 0; ob < 8; ++ob) {
      const float bv = bias[ob * 16 + row];
#pragma unroll
      for (int r = 0; r < 8; ++r)   // D: M = r + hi8, N = row
        out[(size_t)(i0 + r + hi8) * CRED + (ob * 16 + row)] = (__bf16)(acc[ob][r] + bv);
    }
  } else {
    __bf16* out = Vt + (size_t)b * CRED * NPOS;
#pragma unroll
    for (int ob = 0; ob < 8; ++ob) {
      const float bv = bias[ob * 16 + row];
      v8bf pk;
#pragma unroll
      for (int r = 0; r < 8; ++r) pk[r] = (__bf16)(acc[ob][r] + bv);
      // channel-major: row c = ob*16+row, consecutive r -> consecutive n
      *(v8bf*)(out + (size_t)(ob * 16 + row) * NPOS + i0 + hi8) = pk;
    }
  }
}

// ---------------------------------------------------------------------------
// Kernel 4: fused attention with batch-axis softmax.
// One workgroup = 4 waves = one 16-row i-tile; wave w owns batch b=w.
// Per 32-column j-chunk: 8 score WMMAs -> exp -> (issue V loads early) ->
// LDS exchange -> shared denominator over the 4 batches -> P A-frag (bf16)
// -> 8 O-accumulate WMMAs.  V-fragment loads are issued BEFORE the barriers
// so their global latency overlaps the softmax LDS exchange (barriers do not
// drain LOADcnt).  Next chunk's Q rows are prefetched (global_prefetch_b8).
// O written bf16 position-major [B, N, 128].
// grid(392), block(128)
// ---------------------------------------------------------------------------
__global__ void __launch_bounds__(128) attn_fused(const __bf16* __restrict__ Kp,
                                                  const __bf16* __restrict__ Qp,
                                                  const __bf16* __restrict__ Vt,
                                                  __bf16* __restrict__ Ob) {
  __shared__ float e_lds[NBATCH][16][32];
  __shared__ float denom[16][32];

  const int tid  = threadIdx.x;
  const int lane = tid & 31;
  const int wave = tid >> 5;          // == batch index
  const int row  = lane & 15;
  const int hi8  = (lane >> 4) * 8;
  const int i0   = blockIdx.x * 16;
  const float scale = 0.012626906807f;   // 1/sqrt(6272); scores are small
                                         // enough that max-subtraction is
                                         // unnecessary for exp().

  const __bf16* Kb = Kp + (size_t)wave * NPOS * CRED;
  const __bf16* Qb = Qp + (size_t)wave * NPOS * CRED;
  const __bf16* Vb = Vt + (size_t)wave * CRED * NPOS;

  // Resident K A-frags for this i-tile (4 chunks over d=128).
  v16bf kfrag[4];
#pragma unroll
  for (int d = 0; d < 4; ++d)
    kfrag[d] = load_frag(Kb + (size_t)(i0 + row) * CRED + d * 32 + hi8);

  v8f oacc[8];
#pragma unroll
  for (int cb = 0; cb < 8; ++cb) oacc[cb] = (v8f){};

  for (int j0 = 0; j0 < NPOS; j0 += 32) {
    // Prefetch next chunk's Q rows into L0 while we compute this chunk.
    if (j0 + 32 < NPOS) {
      __builtin_prefetch(Qb + (size_t)(j0 + 32 + row) * CRED, 0, 1);
      __builtin_prefetch(Qb + (size_t)(j0 + 48 + row) * CRED, 0, 1);
    }

    // --- scores: load both sub-tiles' Q frags in one clause, then 8 WMMAs ---
    v16bf qfrag[2][4];
#pragma unroll
    for (int jt = 0; jt < 2; ++jt) {
      const __bf16* qrow = Qb + (size_t)(j0 + jt * 16 + row) * CRED + hi8;
#pragma unroll
      for (int d = 0; d < 4; ++d) qfrag[jt][d] = load_frag(qrow + d * 32);
    }
#pragma unroll
    for (int jt = 0; jt < 2; ++jt) {
      v8f s = (v8f){};
#pragma unroll
      for (int d = 0; d < 4; ++d)
        s = wmma_bf16(kfrag[d], qfrag[jt][d], s);   // B[k=d][n=j]
#pragma unroll
      for (int r = 0; r < 8; ++r)               // D: i = r+hi8, j = jt*16+row
        e_lds[wave][r + hi8][jt * 16 + row] = __expf(s[r] * scale);
    }

    // --- issue V fragment loads NOW: latency overlaps the LDS exchange ---
    v16bf vfrag[8];
#pragma unroll
    for (int cb = 0; cb < 8; ++cb)
      vfrag[cb] = load_frag(Vb + (size_t)(cb * 16 + row) * NPOS + j0 + hi8);

    __syncthreads();

    // --- shared denominator across the 4 batches (the softmax axis) ---
    for (int t = tid; t < 16 * 32; t += 128) {
      const int il = t >> 5, jl = t & 31;
      denom[il][jl] = e_lds[0][il][jl] + e_lds[1][il][jl] +
                      e_lds[2][il][jl] + e_lds[3][il][jl];
    }
    __syncthreads();

    // --- P A-frag (16x32, A layout: row = lane%16) ---
    v16bf p;
#pragma unroll
    for (int e = 0; e < 16; ++e) {
      const int jl = ((e >= 8) ? 16 : 0) + hi8 + (e & 7);
      p[e] = (__bf16)(e_lds[wave][row][jl] / denom[row][jl]);
    }

    // --- O += P @ V  (8 column blocks of 16 output channels) ---
#pragma unroll
    for (int cb = 0; cb < 8; ++cb)
      oacc[cb] = wmma_bf16(p, vfrag[cb], oacc[cb]);
    // No trailing barrier needed: next iteration only rewrites this wave's
    // private e_lds[wave] region before the next __syncthreads(), and denom
    // is rewritten only after that barrier.
  }

  __bf16* Os = Ob + (size_t)wave * NPOS * CRED;
#pragma unroll
  for (int cb = 0; cb < 8; ++cb)
#pragma unroll
    for (int r = 0; r < 8; ++r)                 // D: i = r+hi8, c = cb*16+row
      Os[(size_t)(i0 + r + hi8) * CRED + (cb * 16 + row)] = (__bf16)oacc[cb][r];
}

// ---------------------------------------------------------------------------
// Kernel 5: final projection back to 512 channels.
// res[b, o, n] = sum_c O[b, n, c] * r_w[o, c] + r_b[o],  output fp32.
// One wave: 16 positions x 128 output channels.
// grid(392, 4, 4), block(32)
// ---------------------------------------------------------------------------
__global__ void final_proj(const __bf16* __restrict__ Ob, const __bf16* __restrict__ rwb,
                           const float* __restrict__ rb, float* __restrict__ out) {
  const int lane  = threadIdx.x;
  const int row   = lane & 15;
  const int hi8   = (lane >> 4) * 8;
  const int i0    = blockIdx.x * 16;
  const int obase = blockIdx.y * 128;
  const int b     = blockIdx.z;

  const __bf16* Os = Ob + ((size_t)b * NPOS + (i0 + row)) * CRED + hi8;
  v16bf a[4];
#pragma unroll
  for (int d = 0; d < 4; ++d) a[d] = load_frag(Os + d * 32);

#pragma unroll
  for (int ob = 0; ob < 8; ++ob) {
    const int o = obase + ob * 16 + row;
    const __bf16* wrow = rwb + (size_t)o * CRED + hi8;
    // Load all 4 weight frags first, then chain the WMMAs.
    v16bf bf[4];
#pragma unroll
    for (int d = 0; d < 4; ++d) bf[d] = load_frag(wrow + d * 32);
    v8f acc = (v8f){};
#pragma unroll
    for (int d = 0; d < 4; ++d) acc = wmma_bf16(a[d], bf[d], acc);

    const float bv = rb[o];
    float4 lo4 = make_float4(acc[0] + bv, acc[1] + bv, acc[2] + bv, acc[3] + bv);
    float4 hi4 = make_float4(acc[4] + bv, acc[5] + bv, acc[6] + bv, acc[7] + bv);
    float* dst = out + ((size_t)b * CINCH + o) * NPOS + i0 + hi8;  // n = i0+r+hi8
    *(float4*)(dst)     = lo4;
    *(float4*)(dst + 4) = hi4;
  }
}

// ---------------------------------------------------------------------------
// Host launch
// ---------------------------------------------------------------------------
extern "C" void kernel_launch(void* const* d_in, const int* in_sizes, int n_in,
                              void* d_out, int out_size, void* d_ws, size_t ws_size,
                              hipStream_t stream) {
  const float* x  = (const float*)d_in[0];
  const float* kw = (const float*)d_in[1];
  const float* kb = (const float*)d_in[2];
  const float* qw = (const float*)d_in[3];
  const float* qb = (const float*)d_in[4];
  const float* vw = (const float*)d_in[5];
  const float* vb = (const float*)d_in[6];
  const float* rw = (const float*)d_in[7];
  const float* rb = (const float*)d_in[8];
  float* out = (float*)d_out;

  // Workspace layout (bf16 buffers, 256B aligned)
  char* ws = (char*)d_ws;
  size_t off = 0;
  auto take = [&](size_t bytes) -> char* {
    char* p = ws + off;
    off = (off + bytes + 255) & ~(size_t)255;
    return p;
  };
  __bf16* xT  = (__bf16*)take((size_t)NBATCH * NPOS * CINCH * 2);  // ~25.7 MB
  __bf16* Kp  = (__bf16*)take((size_t)NBATCH * NPOS * CRED  * 2);  //  ~6.4 MB
  __bf16* Qp  = (__bf16*)take((size_t)NBATCH * NPOS * CRED  * 2);
  __bf16* Vt  = (__bf16*)take((size_t)NBATCH * CRED * NPOS  * 2);
  __bf16* Ob  = (__bf16*)take((size_t)NBATCH * NPOS * CRED  * 2);
  __bf16* kwb = (__bf16*)take((size_t)CRED * CINCH * 2);
  __bf16* qwb = (__bf16*)take((size_t)CRED * CINCH * 2);
  __bf16* vwb = (__bf16*)take((size_t)CRED * CINCH * 2);
  __bf16* rwb = (__bf16*)take((size_t)CINCH * CRED * 2);
  (void)ws_size; (void)in_sizes; (void)n_in; (void)out_size;

  // 1. weights -> bf16
  convert_weights<<<(CRED * CINCH + 255) / 256, 256, 0, stream>>>(
      kw, qw, vw, rw, kwb, qwb, vwb, rwb);

  // 2. x -> xT bf16
  transpose_convert_x<<<dim3(NPOS / 32, CINCH / 32, NBATCH), dim3(32, 8), 0, stream>>>(x, xT);

  // 3. K/Q/V projections (WMMA)
  proj_kqv<<<dim3(NPOS / 16, 3, NBATCH), dim3(32), 0, stream>>>(
      xT, kwb, qwb, vwb, kb, qb, vb, Kp, Qp, Vt);

  // 4. fused attention with batch-axis softmax (WMMA)
  attn_fused<<<dim3(NPOS / 16), dim3(128), 0, stream>>>(Kp, Qp, Vt, Ob);

  // 5. final 512-channel projection (WMMA)
  final_proj<<<dim3(NPOS / 16, CINCH / 128, NBATCH), dim3(32), 0, stream>>>(Ob, rwb, rb, out);
}
